// LinearAttention_25451976196659
// MI455X (gfx1250) — matrix operand order repaired and checked by reading, not compiled, not run on hardware
//
#include <hip/hip_runtime.h>
#include <math.h>

// Problem constants (from reference)
#define SEQ   2048
#define BATCH 2
#define DM    1024
#define NH    16
#define DH    64
#define ROWS  (SEQ * BATCH)   // 4096
#define CPAD  68              // LDS row stride (floats): 68 % 64 == 4 -> conflict-free columns

typedef __attribute__((ext_vector_type(16))) __bf16 bf16x16;
typedef __attribute__((ext_vector_type(8)))  __bf16 bf16x8;
typedef __attribute__((ext_vector_type(8)))  float  f32x8;
typedef __attribute__((ext_vector_type(2)))  float  f32x2;

static __device__ inline f32x8 wmma_bf16(bf16x16 a, bf16x16 b, f32x8 c) {
  // D(16x16,f32) = A(16x32,bf16) * B(32x16,bf16) + C
  return __builtin_amdgcn_wmma_f32_16x16x32_bf16(false, a, false, b, (short)0, c, false, false);
}
static __device__ inline f32x8 wmma_f32k4(f32x2 a, f32x2 b, f32x8 c) {
  // D(16x16,f32) = A(16x4,f32) * B(4x16,f32) + C   (exact f32 path)
  return __builtin_amdgcn_wmma_f32_16x16x4_f32(false, a, false, b, (short)0, c, false, false);
}

static __device__ inline float softplus_f(float x) {
  return (x > 15.0f) ? x : log1pf(expf(x));
}

// Async global->LDS DMA (ASYNCcnt-tracked), GVS mode:
//   LDS[lds_addr + byte] = MEM[saddr + voff + byte], 16 bytes per lane.
static __device__ inline void async_ld_b128(unsigned int lds_addr, unsigned int voff,
                                            unsigned long long sbase) {
  asm volatile("global_load_async_to_lds_b128 %0, %1, %2"
               :: "v"(lds_addr), "v"(voff), "s"(sbase) : "memory");
}
static __device__ inline void wait_async0() {
  asm volatile("s_wait_asynccnt 0x0" ::: "memory");
}

// ---------------------------------------------------------------------------
// f32 -> bf16 conversion (grid-stride)
// ---------------------------------------------------------------------------
__global__ __launch_bounds__(256)
void la_cvt_bf16(const float* __restrict__ src, __bf16* __restrict__ dst, int n) {
  int i = blockIdx.x * blockDim.x + threadIdx.x;
  int stride = gridDim.x * blockDim.x;
  for (; i < n; i += stride) dst[i] = (__bf16)src[i];
}

// ---------------------------------------------------------------------------
// Fragment loaders (contiguous k order, same for A and B; not the ISA table order)
//   A (16-bit, 16x32):  lane L: row m=L&15; half=L>>4
//       element e -> K = 16*half + e              (32B contiguous)
//   B (16-bit, 32x16):  lane L: col n=L&15; element e -> K = 16*half + e
//       -> 32B contiguous in a row of W (since B[k][n] = W[n][k])
// ---------------------------------------------------------------------------
static __device__ inline bf16x16 load_a_frag(const __bf16* __restrict__ rowptr,
                                             int kbase, int halfid) {
  bf16x8 lo = *(const bf16x8*)(rowptr + kbase + halfid * 16);
  bf16x8 hi = *(const bf16x8*)(rowptr + kbase + halfid * 16 + 8);
  return __builtin_shufflevector(lo, hi, 0,1,2,3,4,5,6,7,8,9,10,11,12,13,14,15);
}

// ---------------------------------------------------------------------------
// Fused QKV projection GEMM (bf16 WMMA, f32 accum) + temperature + softplus.
//   y = x @ W^T + b ; q,k = softplus(y / (8*exp(beta_h))) ; v = y
// Register blocking: each wave computes a 16x32 tile for Q, K and V
// (6 WMMA per A-fragment). Grid: 1024 blocks x 256 threads.
// ---------------------------------------------------------------------------
__global__ __launch_bounds__(256)
void la_qkv_gemm(const __bf16* __restrict__ xb,
                 const __bf16* __restrict__ Wq, const __bf16* __restrict__ Wk,
                 const __bf16* __restrict__ Wv,
                 const float* __restrict__ bq, const float* __restrict__ bk,
                 const float* __restrict__ bv,
                 const float* __restrict__ beta,
                 float* __restrict__ Qg, float* __restrict__ Kg, float* __restrict__ Vg) {
  const int lane   = threadIdx.x & 31;
  const int wave   = threadIdx.x >> 5;
  const int gwave  = blockIdx.x * 8 + wave;      // 8192 wave-tiles
  const int ngrp   = gwave & 31;                 // 32 groups of 2 N-tiles
  const int mtile  = gwave >> 5;                 // 256 M tiles
  const int mn     = lane & 15;
  const int halfid = lane >> 4;

  const __bf16* pA = xb + (size_t)(mtile * 16 + mn) * DM;
  const __bf16* pB[6];
#pragma unroll
  for (int j = 0; j < 2; ++j) {
    size_t rowoff = (size_t)(ngrp * 32 + j * 16 + mn) * DM + halfid * 16;
    pB[j * 3 + 0] = Wq + rowoff;
    pB[j * 3 + 1] = Wk + rowoff;
    pB[j * 3 + 2] = Wv + rowoff;
  }

  f32x8 acc[6] = {};
  for (int kk = 0; kk < DM; kk += 32) {
    if (kk + 128 < DM) {   // global_prefetch_b8, near scope
      __builtin_prefetch(pA + kk + 128, 0, 3);
#pragma unroll
      for (int j = 0; j < 6; ++j) __builtin_prefetch(pB[j] + kk + 128, 0, 3);
    }
    bf16x16 afrag = load_a_frag(pA, kk, halfid);
#pragma unroll
    for (int j = 0; j < 6; ++j) {
      bf16x16 bf = *(const bf16x16*)(pB[j] + kk);
      acc[j] = wmma_bf16(afrag, bf, acc[j]);
    }
  }

  // Epilogue: per-head temperature + softplus, scatter to (b,h,s,dh)
#pragma unroll
  for (int j = 0; j < 2; ++j) {
    const int   col = ngrp * 32 + j * 16 + mn;
    const int   h   = col >> 6;                       // head = col / 64
    const float inv = 1.0f / (8.0f * expf(beta[h]));  // sqrt(64)*exp(beta)
    const float bqc = bq[col], bkc = bk[col], bvc = bv[col];
#pragma unroll
    for (int v = 0; v < 8; ++v) {
      int r  = mtile * 16 + halfid * 8 + v;           // flat (s,b) row
      int s  = r >> 1;
      int bb = r & 1;
      size_t o = ((size_t)(bb * NH + h) * SEQ + s) * DH + (col & (DH - 1));
      Qg[o] = softplus_f((acc[j * 3 + 0][v] + bqc) * inv);
      Kg[o] = softplus_f((acc[j * 3 + 1][v] + bkc) * inv);
      Vg[o] = acc[j * 3 + 2][v] + bvc;
    }
  }
}

// ---------------------------------------------------------------------------
// Chunked causal linear attention, one workgroup per (b,h).
//   per chunk c (64 rows):  S = tril(Qc Kc^T)
//                           den_i = rowsum(S)_i + Qc_i . kn
//                           O = S @ Vc + Qc @ State ;  out = O / den
//                           State += Kc^T @ Vc ;  kn += colsum(Kc)
// Q/K/V chunks staged into LDS via GLOBAL_LOAD_ASYNC_TO_LDS_B128 (ASYNCcnt).
// All chunk matmuls via exact-f32 V_WMMA_F32_16X16X4_F32.
// ---------------------------------------------------------------------------
__global__ __launch_bounds__(256)
void la_attn(const float* __restrict__ Qg, const float* __restrict__ Kg,
             const float* __restrict__ Vg, __bf16* __restrict__ attnb) {
  extern __shared__ float smem[];
  float* Qs = smem;                 // 64 x CPAD
  float* Ks = Qs + 64 * CPAD;
  float* Vs = Ks + 64 * CPAD;
  float* Ss = Vs + 64 * CPAD;
  float* St = Ss + 64 * CPAD;       // running state (dk x dv)
  float* kn = St + 64 * CPAD;       // running key sum (64)
  float* den = kn + 64;             // per-row denominator (64)

  const int tid    = threadIdx.x;
  const int lane   = tid & 31;
  const int wave   = tid >> 5;
  const int mn     = lane & 15;
  const int halfid = lane >> 4;
  const int bh     = blockIdx.x;    // b*16 + h
  const int bb     = bh >> 4;
  const int h      = bh & 15;
  const size_t base = (size_t)bh * SEQ * DH;

  // LDS byte addresses (low 32 bits of generic shared pointer = LDS offset)
  const unsigned int qs_lds = (unsigned int)(uintptr_t)Qs;
  const unsigned int ks_lds = (unsigned int)(uintptr_t)Ks;
  const unsigned int vs_lds = (unsigned int)(uintptr_t)Vs;
  const unsigned long long qbase = (unsigned long long)(uintptr_t)(Qg + base);
  const unsigned long long kbase = (unsigned long long)(uintptr_t)(Kg + base);
  const unsigned long long vbase = (unsigned long long)(uintptr_t)(Vg + base);

  for (int i = tid; i < 64 * CPAD; i += 256) St[i] = 0.0f;
  if (tid < 64) kn[tid] = 0.0f;
  __syncthreads();

  for (int c = 0; c < SEQ / 64; ++c) {
    // Stage Qc, Kc, Vc into LDS: 16B per lane per op, async DMA
    for (int idx = tid; idx < 64 * 16; idx += 256) {     // 1024 float4 groups
      int row = idx >> 4, c4 = idx & 15;
      unsigned int goff = (unsigned int)(((c * 64 + row) * DH + c4 * 4) * 4);
      unsigned int loff = (unsigned int)((row * CPAD + c4 * 4) * 4);
      async_ld_b128(qs_lds + loff, goff, qbase);
      async_ld_b128(ks_lds + loff, goff, kbase);
      async_ld_b128(vs_lds + loff, goff, vbase);
    }
    wait_async0();
    __syncthreads();

    // S = tril(Qc Kc^T): 16 tiles, 2 per wave
#pragma unroll
    for (int tt = 0; tt < 2; ++tt) {
      int t = wave + tt * 8, ti = t >> 2, tj = t & 3;
      f32x8 acc = {};
      for (int kk = 0; kk < 16; ++kk) {
        int kb = kk * 4 + halfid * 2;
        f32x2 a = *(const f32x2*)&Qs[(ti * 16 + mn) * CPAD + kb];   // ds_load_b64
        f32x2 b = *(const f32x2*)&Ks[(tj * 16 + mn) * CPAD + kb];
        acc = wmma_f32k4(a, b, acc);
      }
#pragma unroll
      for (int v = 0; v < 8; ++v) {
        int il = ti * 16 + halfid * 8 + v, jl = tj * 16 + mn;
        Ss[il * CPAD + jl] = (jl <= il) ? acc[v] : 0.0f;   // causal mask
      }
    }
    __syncthreads();

    // den_i = rowsum(S)_i + Qc_i . kn   (uses OLD kn)
    if (tid < 64) {
      float d = 0.0f;
      for (int j = 0; j < 64; ++j) d += Ss[tid * CPAD + j];
      float qk = 0.0f;
      for (int dd = 0; dd < 64; ++dd) qk += Qs[tid * CPAD + dd] * kn[dd];
      den[tid] = d + qk;
    }
    __syncthreads();

    // O = S @ Vc + Qc @ State ; out = O / den  -> bf16 (s,b,d)
#pragma unroll
    for (int tt = 0; tt < 2; ++tt) {
      int t = wave + tt * 8, ti = t >> 2, tj = t & 3;
      f32x8 acc = {};
      for (int kk = 0; kk < 16; ++kk) {
        int kb = kk * 4 + halfid * 2;
        f32x2 a = *(const f32x2*)&Ss[(ti * 16 + mn) * CPAD + kb];
        f32x2 b = { Vs[kb * CPAD + tj * 16 + mn], Vs[(kb + 1) * CPAD + tj * 16 + mn] };
        acc = wmma_f32k4(a, b, acc);
      }
      for (int kk = 0; kk < 16; ++kk) {
        int kb = kk * 4 + halfid * 2;
        f32x2 a = *(const f32x2*)&Qs[(ti * 16 + mn) * CPAD + kb];
        f32x2 b = { St[kb * CPAD + tj * 16 + mn], St[(kb + 1) * CPAD + tj * 16 + mn] };
        acc = wmma_f32k4(a, b, acc);
      }
#pragma unroll
      for (int v = 0; v < 8; ++v) {
        int il = ti * 16 + halfid * 8 + v;
        float o = acc[v] / den[il];
        int sg = c * 64 + il;
        int r  = sg * BATCH + bb;
        attnb[(size_t)r * DM + h * DH + tj * 16 + mn] = (__bf16)o;
      }
    }
    __syncthreads();   // all reads of State complete before update

    // kn += colsum(Kc)
    if (tid < 64) {
      float a = kn[tid];
      for (int p = 0; p < 64; ++p) a += Ks[p * CPAD + tid];
      kn[tid] = a;
    }
    // State += Kc^T @ Vc  (A[d][pos] = Kc[pos][d])
#pragma unroll
    for (int tt = 0; tt < 2; ++tt) {
      int t = wave + tt * 8, ti = t >> 2, tj = t & 3;
      f32x8 acc = {};
#pragma unroll
      for (int v = 0; v < 8; ++v)
        acc[v] = St[(ti * 16 + halfid * 8 + v) * CPAD + tj * 16 + mn];
      for (int kk = 0; kk < 16; ++kk) {
        int kb = kk * 4 + halfid * 2;
        f32x2 a = { Ks[kb * CPAD + ti * 16 + mn], Ks[(kb + 1) * CPAD + ti * 16 + mn] };
        f32x2 b = { Vs[kb * CPAD + tj * 16 + mn], Vs[(kb + 1) * CPAD + tj * 16 + mn] };
        acc = wmma_f32k4(a, b, acc);
      }
#pragma unroll
      for (int v = 0; v < 8; ++v)
        St[(ti * 16 + halfid * 8 + v) * CPAD + tj * 16 + mn] = acc[v];
    }
    __syncthreads();
  }
}

// ---------------------------------------------------------------------------
// Output projection: out = attn @ Wo^T + bo   (bf16 WMMA, f32 out = d_out)
// Register blocking: each wave computes a 16x64 tile (4 WMMA per A-fragment).
// ---------------------------------------------------------------------------
__global__ __launch_bounds__(256)
void la_out_gemm(const __bf16* __restrict__ A, const __bf16* __restrict__ Wo,
                 const float* __restrict__ bo, float* __restrict__ out) {
  const int lane   = threadIdx.x & 31;
  const int wave   = threadIdx.x >> 5;
  const int gwave  = blockIdx.x * 8 + wave;      // 4096 wave-tiles
  const int ngrp   = gwave & 15;                 // 16 groups of 4 N-tiles
  const int mtile  = gwave >> 4;                 // 256 M tiles
  const int mn     = lane & 15;
  const int halfid = lane >> 4;

  const __bf16* pA = A + (size_t)(mtile * 16 + mn) * DM;
  const __bf16* pB[4];
#pragma unroll
  for (int j = 0; j < 4; ++j)
    pB[j] = Wo + (size_t)(ngrp * 64 + j * 16 + mn) * DM + halfid * 16;

  f32x8 acc[4] = {};
  for (int kk = 0; kk < DM; kk += 32) {
    if (kk + 128 < DM) {
      __builtin_prefetch(pA + kk + 128, 0, 3);
#pragma unroll
      for (int j = 0; j < 4; ++j) __builtin_prefetch(pB[j] + kk + 128, 0, 3);
    }
    bf16x16 afrag = load_a_frag(pA, kk, halfid);
#pragma unroll
    for (int j = 0; j < 4; ++j) {
      bf16x16 bf = *(const bf16x16*)(pB[j] + kk);
      acc[j] = wmma_bf16(afrag, bf, acc[j]);
    }
  }
#pragma unroll
  for (int j = 0; j < 4; ++j) {
    const int col = ngrp * 64 + j * 16 + mn;
    const float boc = bo[col];
#pragma unroll
    for (int v = 0; v < 8; ++v) {
      int r = mtile * 16 + halfid * 8 + v;
      out[(size_t)r * DM + col] = acc[j][v] + boc;
    }
  }
}

// ---------------------------------------------------------------------------
// Launch
// ---------------------------------------------------------------------------
extern "C" void kernel_launch(void* const* d_in, const int* in_sizes, int n_in,
                              void* d_out, int out_size, void* d_ws, size_t ws_size,
                              hipStream_t stream) {
  const float* x    = (const float*)d_in[0];
  const float* beta = (const float*)d_in[1];
  const float* Wq   = (const float*)d_in[2];
  const float* bq   = (const float*)d_in[3];
  const float* Wk   = (const float*)d_in[4];
  const float* bk   = (const float*)d_in[5];
  const float* Wv   = (const float*)d_in[6];
  const float* bv   = (const float*)d_in[7];
  const float* Wo   = (const float*)d_in[8];
  const float* bo   = (const float*)d_in[9];
  float* out = (float*)d_out;

  // Workspace layout (~72 MB)
  __bf16* xb    = (__bf16*)d_ws;                       // 4096x1024 bf16
  __bf16* Wqb   = xb  + (size_t)ROWS * DM;             // 1024x1024 bf16
  __bf16* Wkb   = Wqb + (size_t)DM * DM;
  __bf16* Wvb   = Wkb + (size_t)DM * DM;
  __bf16* Wob   = Wvb + (size_t)DM * DM;
  float*  Qg    = (float*)(Wob + (size_t)DM * DM);     // (b,h,s,dh) f32
  float*  Kg    = Qg + (size_t)ROWS * DM;
  float*  Vg    = Kg + (size_t)ROWS * DM;
  __bf16* attnb = (__bf16*)(Vg + (size_t)ROWS * DM);   // 4096x1024 bf16

  la_cvt_bf16<<<2048, 256, 0, stream>>>(x,  xb,  ROWS * DM);
  la_cvt_bf16<<<1024, 256, 0, stream>>>(Wq, Wqb, DM * DM);
  la_cvt_bf16<<<1024, 256, 0, stream>>>(Wk, Wkb, DM * DM);
  la_cvt_bf16<<<1024, 256, 0, stream>>>(Wv, Wvb, DM * DM);
  la_cvt_bf16<<<1024, 256, 0, stream>>>(Wo, Wob, DM * DM);

  la_qkv_gemm<<<1024, 256, 0, stream>>>(xb, Wqb, Wkb, Wvb, bq, bk, bv, beta, Qg, Kg, Vg);

  size_t smem_bytes = (size_t)(5 * 64 * CPAD + 128) * sizeof(float);  // 87,552 B
  la_attn<<<NH * BATCH, 256, smem_bytes, stream>>>(Qg, Kg, Vg, attnb);

  la_out_gemm<<<512, 256, 0, stream>>>(attnb, Wob, bo, out);
}
